// LSTM_15625091023103
// MI455X (gfx1250) — compile-verified, matrix operand order checked by reading, not run on hardware
//
#include <hip/hip_runtime.h>
#include <hip/hip_bf16.h>

#define DEVINL __device__ __forceinline__

typedef __bf16 bf16_t;
typedef bf16_t v16bf __attribute__((ext_vector_type(16)));
typedef float  v8f   __attribute__((ext_vector_type(8)));
typedef unsigned int u32x4 __attribute__((ext_vector_type(4)));
typedef float f32x4 __attribute__((ext_vector_type(4)));

constexpr int kB = 256;    // batch
constexpr int kT = 2048;   // sequence length
constexpr int kI = 128;    // input dim
constexpr int kH = 128;    // hidden dim
constexpr int kG = 4 * kH; // 512 gate columns

// ---------- workspace layout (bytes) ----------
constexpr size_t kOffBias = 0;
constexpr size_t kOffWi   = 2048;
constexpr size_t kOffWh   = kOffWi + (size_t)kG * kI * sizeof(bf16_t);
constexpr size_t kOffXi   = kOffWh + (size_t)kG * kI * sizeof(bf16_t);

DEVINL float fast_sigmoid(float x) { return 1.0f / (1.0f + __expf(-x)); }
DEVINL float fast_tanh(float x)    { return 1.0f - 2.0f / (__expf(2.0f * x) + 1.0f); }

// Load a 16-bit B-matrix tile (32x16, K x N) for v_wmma_*_16x16x32 from a
// row-major (G x 128) bf16 weight array: B[k][n] = W[g][k].
DEVINL v16bf load_Btile(const bf16_t* __restrict__ W, int g, int kt, int khalf) {
  const bf16_t* p = W + (long)g * kI + kt * 32 + khalf;
  union { v16bf v; u32x4 q[2]; } u;
  u.q[0] = *(const u32x4*)(p);
  u.q[1] = *(const u32x4*)(p + 16);
  return u.v;
}

// Issue an async copy of a 32 KB slab (16 x 512 f32, contiguous) from global
// to LDS: 8 x global_load_async_to_lds_b128 per thread (ASYNCcnt += 8).
// INST_OFFSET is added to both the LDS and global addresses (ISA 10.x async
// pseudocode), so one LDS-offset VGPR + one 64-bit global VGPR pair suffice.
DEVINL void issue_xi_async(const float* __restrict__ gsrc, const float* lds_dst, int tid) {
  unsigned lds = (unsigned)(unsigned long long)lds_dst + (unsigned)tid * 16u;
  unsigned long long g = (unsigned long long)gsrc + (unsigned long long)tid * 16ull;
  asm volatile("global_load_async_to_lds_b128 %0, %1, off offset:0"     :: "v"(lds), "v"(g) : "memory");
  asm volatile("global_load_async_to_lds_b128 %0, %1, off offset:4096"  :: "v"(lds), "v"(g) : "memory");
  asm volatile("global_load_async_to_lds_b128 %0, %1, off offset:8192"  :: "v"(lds), "v"(g) : "memory");
  asm volatile("global_load_async_to_lds_b128 %0, %1, off offset:12288" :: "v"(lds), "v"(g) : "memory");
  asm volatile("global_load_async_to_lds_b128 %0, %1, off offset:16384" :: "v"(lds), "v"(g) : "memory");
  asm volatile("global_load_async_to_lds_b128 %0, %1, off offset:20480" :: "v"(lds), "v"(g) : "memory");
  asm volatile("global_load_async_to_lds_b128 %0, %1, off offset:24576" :: "v"(lds), "v"(g) : "memory");
  asm volatile("global_load_async_to_lds_b128 %0, %1, off offset:28672" :: "v"(lds), "v"(g) : "memory");
}

// ---------------------------------------------------------------------------
// Kernel 0: weight conversion fp32 -> bf16, fused bias
// ---------------------------------------------------------------------------
__global__ void lstm_prep_kernel(const float* __restrict__ Wi, const float* __restrict__ bi,
                                 const float* __restrict__ Wh, const float* __restrict__ bh,
                                 float* __restrict__ bias, bf16_t* __restrict__ Wi_bf,
                                 bf16_t* __restrict__ Wh_bf) {
  int i = blockIdx.x * blockDim.x + threadIdx.x;
  if (i < kG * kI) {
    Wi_bf[i] = (bf16_t)Wi[i];
    Wh_bf[i] = (bf16_t)Wh[i];
  }
  if (i < kG) bias[i] = bi[i] + bh[i];
}

// ---------------------------------------------------------------------------
// Kernel 1: xi[t][b][g] = x[b][t][:] @ Wi[g][:] + (bi+bh)[g]
// ---------------------------------------------------------------------------
__global__ void __launch_bounds__(256) lstm_xi_gemm_kernel(
    const float* __restrict__ x, const bf16_t* __restrict__ Wi_bf,
    const float* __restrict__ bias, float* __restrict__ xi) {
  const int lane  = threadIdx.x & 31;
  const int wave  = threadIdx.x >> 5;
  const int mtile = blockIdx.x * 8 + wave;
  const long row0 = (long)mtile * 16;           // flattened (b,t) row
  const int  b    = (int)(row0 / kT);
  const int  t0   = (int)(row0 % kT);           // 16 consecutive t, same b
  const int  col  = lane & 15;
  const int  hi   = lane >> 4;
  const int  khalf = hi * 8;
  const int  mbase = hi * 8;

  // A panel: 16 rows x K=128, converted fp32 -> bf16, A-matrix VGPR layout.
  v16bf A[4];
  {
    const float* xrow = x + (row0 + col) * (long)kI;
#pragma unroll
    for (int kt = 0; kt < 4; ++kt) {
      const float* p = xrow + kt * 32 + khalf;
      f32x4 f0 = *(const f32x4*)(p);
      f32x4 f1 = *(const f32x4*)(p + 4);
      f32x4 f2 = *(const f32x4*)(p + 16);
      f32x4 f3 = *(const f32x4*)(p + 20);
      v16bf a;
#pragma unroll
      for (int j = 0; j < 4; ++j) {
        a[j]      = (bf16_t)f0[j];
        a[4 + j]  = (bf16_t)f1[j];
        a[8 + j]  = (bf16_t)f2[j];
        a[12 + j] = (bf16_t)f3[j];
      }
      A[kt] = a;
    }
  }

#pragma unroll 4
  for (int nt = 0; nt < 32; ++nt) {
    const int g = nt * 16 + col;
    const float bv = bias[g];
    v8f acc;
#pragma unroll
    for (int r = 0; r < 8; ++r) acc[r] = bv;
#pragma unroll
    for (int kt = 0; kt < 4; ++kt) {
      v16bf Bt = load_Btile(Wi_bf, g, kt, khalf);
      acc = __builtin_amdgcn_wmma_f32_16x16x32_bf16(false, A[kt], false, Bt,
                                                    (short)0, acc, false, false);
    }
#pragma unroll
    for (int r = 0; r < 8; ++r) {
      const int m = mbase + r;
      xi[((long)(t0 + m) * kB + b) * kG + g] = acc[r];
    }
  }
}

// ---------------------------------------------------------------------------
// Kernel 2: sequential scan with async double-buffered xi staging.
// One block per 16 batch rows; 8 waves split the 512 gate columns (64 each).
// Wh tiles live in VGPRs across all 2048 steps; h (bf16) / c (f32) in LDS.
// ---------------------------------------------------------------------------
__global__ void __launch_bounds__(256, 1) lstm_scan_kernel(
    const float* __restrict__ xi, const bf16_t* __restrict__ Wh_bf,
    float* __restrict__ out, float* __restrict__ hf, float* __restrict__ cf) {
  __shared__ float  xi_buf[2][16][kG];  // 64 KB double-buffered xi slabs
  __shared__ float  lin_s[16][kG];      // 32 KB gate pre-activations
  __shared__ float  c_s[16][kH];        // 8 KB cell state
  __shared__ bf16_t h_s[16][kH];        // 4 KB hidden state (next A operand)

  const int tid   = threadIdx.x;
  const int lane  = tid & 31;
  const int wave  = tid >> 5;
  const int b0    = blockIdx.x * 16;
  const int col   = lane & 15;
  const int hi    = lane >> 4;
  const int khalf = hi * 8;
  const int mbase = hi * 8;
  const int gbase = wave * 64;

  for (int e = tid; e < 16 * kH; e += 256) {
    c_s[e >> 7][e & 127] = 0.0f;
    h_s[e >> 7][e & 127] = (bf16_t)0.0f;
  }

  // Step-invariant Wh B-tiles: [ntile][ktile], 128 VGPRs.
  v16bf Bh[4][4];
#pragma unroll
  for (int nt = 0; nt < 4; ++nt)
#pragma unroll
    for (int kt = 0; kt < 4; ++kt)
      Bh[nt][kt] = load_Btile(Wh_bf, gbase + nt * 16 + col, kt, khalf);

  // Kick off async copy of slab t=0.
  issue_xi_async(xi + (long)b0 * kG, &xi_buf[0][0][0], tid);

#pragma unroll 1
  for (int t = 0; t < kT; ++t) {
    const int cur = t & 1;
    if (t + 1 < kT) {
      // Overlap: stage slab t+1 while computing step t.
      issue_xi_async(xi + ((long)(t + 1) * kB + b0) * kG, &xi_buf[cur ^ 1][0][0], tid);
      asm volatile("s_wait_asynccnt 0x8" ::: "memory");  // slab t done, t+1 in flight
    } else {
      asm volatile("s_wait_asynccnt 0x0" ::: "memory");  // drain final slab
    }
    __syncthreads();  // slab t visible to all waves; h_s from step t-1 visible

    // A tiles (16x128 bf16) from LDS-resident h.
    v16bf A[4];
#pragma unroll
    for (int kt = 0; kt < 4; ++kt) {
      const bf16_t* p = &h_s[col][kt * 32 + khalf];
      union { v16bf v; u32x4 q[2]; } u;
      u.q[0] = *(const u32x4*)(p);
      u.q[1] = *(const u32x4*)(p + 16);
      A[kt] = u.v;
    }

#pragma unroll
    for (int nt = 0; nt < 4; ++nt) {
      const int g = gbase + nt * 16 + col;
      v8f acc;
#pragma unroll
      for (int r = 0; r < 8; ++r) acc[r] = xi_buf[cur][mbase + r][g];
#pragma unroll
      for (int kt = 0; kt < 4; ++kt)
        acc = __builtin_amdgcn_wmma_f32_16x16x32_bf16(false, A[kt], false, Bh[nt][kt],
                                                      (short)0, acc, false, false);
#pragma unroll
      for (int r = 0; r < 8; ++r) lin_s[mbase + r][g] = acc[r];
    }
    __syncthreads();

    // Gate math: 2048 elements, 8 per thread (one row, 8 consecutive cols).
    {
      const int row = tid >> 4;
      const int j0  = (tid & 15) * 8;
#pragma unroll
      for (int jj = 0; jj < 8; ++jj) {
        const int j   = j0 + jj;
        const float fg = fast_sigmoid(lin_s[row][j]);
        const float ig = fast_sigmoid(lin_s[row][kH + j]);
        const float og = fast_sigmoid(lin_s[row][2 * kH + j]);
        const float cc = fast_tanh(lin_s[row][3 * kH + j]);
        const float cn = c_s[row][j] * fg + ig * cc;
        c_s[row][j] = cn;
        const float hn = og * fast_tanh(cn);
        h_s[row][j] = (bf16_t)hn;
        out[((long)(b0 + row) * kT + t) * kH + j] = hn;
        if (t == kT - 1) {
          hf[(b0 + row) * kH + j] = hn;
          cf[(b0 + row) * kH + j] = cn;
        }
      }
    }
    // Next iteration's pre-GEMM barrier orders h_s/c_s writes vs reads.
  }
}

// ---------------------------------------------------------------------------
extern "C" void kernel_launch(void* const* d_in, const int* in_sizes, int n_in,
                              void* d_out, int out_size, void* d_ws, size_t ws_size,
                              hipStream_t stream) {
  (void)in_sizes; (void)n_in; (void)out_size; (void)ws_size;
  const float* x  = (const float*)d_in[0];
  const float* Wi = (const float*)d_in[1];
  const float* bi = (const float*)d_in[2];
  const float* Wh = (const float*)d_in[3];
  const float* bh = (const float*)d_in[4];
  float* out = (float*)d_out;

  char* ws = (char*)d_ws;
  float*  bias  = (float*)(ws + kOffBias);
  bf16_t* Wi_bf = (bf16_t*)(ws + kOffWi);
  bf16_t* Wh_bf = (bf16_t*)(ws + kOffWh);
  float*  xi    = (float*)(ws + kOffXi);   // needs ~1 GB of workspace

  float* hf = out + (long)kB * kT * kH;
  float* cf = hf + (long)kB * kH;

  lstm_prep_kernel<<<256, 256, 0, stream>>>(Wi, bi, Wh, bh, bias, Wi_bf, Wh_bf);

  // M = B*T = 524288 rows -> 32768 M-tiles -> 4096 blocks of 8 waves.
  lstm_xi_gemm_kernel<<<4096, 256, 0, stream>>>(x, Wi_bf, bias, xi);

  // One block per 16 batch rows.
  lstm_scan_kernel<<<kB / 16, 256, 0, stream>>>(xi, Wh_bf, out, hf, cf);
}
